// HydraAttention_4569845202973
// MI455X (gfx1250) — matrix operand order repaired and checked by reading, not compiled, not run on hardware
//
#include <hip/hip_runtime.h>

// ---------------------------------------------------------------------------
// HydraAttention for MI455X (gfx1250, wave32, WMMA bf16 16x16x32, f32 accum)
//   T=2048 B=8 E=1024 H=16 hd=64  -> BH=128, M=16384
// GEMMs use double-buffered GLOBAL_LOAD_ASYNC_TO_LDS_B128 (ASYNCcnt) so the
// global->LDS DMA overlaps the WMMA pipeline.
// Scratch layout (d_ws), ~168 MB total:
//   [0]           qkv  bf16  [16384 x 3072]    100663296 B
//   [+100663296]  qbf  bf16  [16384 x 1024]     33554432 B
//   [+134217728]  attn bf16  [16384 x 1024]     33554432 B
//   [+167772160]  wqkv bf16  [ 3072 x 1024]      6291456 B
//   [+174063616]  wout bf16  [ 1024 x 1024]      2097152 B
// ---------------------------------------------------------------------------

#define T_LEN 2048
#define BSZ   8
#define EMB   1024
#define NHEAD 16
#define HD    64
#define BH    128
#define MROWS 16384
#define QKVN  3072

typedef __attribute__((ext_vector_type(16))) __bf16 bf16x16;
typedef __attribute__((ext_vector_type(8)))  float  f32x8;

union Frag { bf16x16 v; uint4 q[2]; };

__device__ __forceinline__ unsigned short bf16_rn(float f) {
  union { float f; unsigned int u; } c; c.f = f;
  unsigned int u = c.u + 0x7FFFu + ((c.u >> 16) & 1u);  // round-to-nearest-even
  return (unsigned short)(u >> 16);
}
__device__ __forceinline__ float bf16_f32(unsigned short h) {
  union { unsigned int u; float f; } c; c.u = ((unsigned int)h) << 16;
  return c.f;
}

// LDS byte address of a __shared__ object (flat address truncates to LDS offset)
__device__ __forceinline__ unsigned lds_off(const void* p) {
  return (unsigned)(size_t)p;
}

// Async DMA: 32 bytes global -> LDS (two B128 ops, offset applies to both sides).
// Tracked by ASYNCcnt; consumer must s_wait_asynccnt + barrier before ds reads.
__device__ __forceinline__ void async_copy32(unsigned lds, unsigned goff,
                                             const void* base) {
  asm volatile(
      "global_load_async_to_lds_b128 %0, %1, %2\n\t"
      "global_load_async_to_lds_b128 %0, %1, %2 offset:16"
      :: "v"(lds), "v"(goff), "s"((unsigned long long)(size_t)base)
      : "memory");
}
__device__ __forceinline__ void wait_async0() {
  asm volatile("s_wait_asynccnt 0x0" ::: "memory");
}

// ---------------------------------------------------------------------------
// fp32 -> bf16 bulk convert (n multiple of 4)
// ---------------------------------------------------------------------------
__global__ __launch_bounds__(256) void cvt_f32_bf16_kernel(
    const float* __restrict__ src, unsigned short* __restrict__ dst, int n4) {
  int i = blockIdx.x * blockDim.x + threadIdx.x;
  int stride = gridDim.x * blockDim.x;
  for (; i < n4; i += stride) {
    float4 f = ((const float4*)src)[i];
    union { unsigned short u[4]; unsigned long long ll; } o;
    o.u[0] = bf16_rn(f.x); o.u[1] = bf16_rn(f.y);
    o.u[2] = bf16_rn(f.z); o.u[3] = bf16_rn(f.w);
    ((unsigned long long*)dst)[i] = o.ll;
  }
}

// ---------------------------------------------------------------------------
// C[M,N] = A[M,K](bf16,rm) * W[N,K](bf16,rm)^T + bias[N]
// 128x128 block, 8 waves of 64(M)x32(N), K-step 32.
// Double-buffered async global->LDS staging; LDS row stride 40 ushorts.
// ---------------------------------------------------------------------------
template <bool BF16_OUT>
__global__ __launch_bounds__(256) void gemm_bf16_kernel(
    const unsigned short* __restrict__ A, const unsigned short* __restrict__ W,
    const float* __restrict__ bias, void* __restrict__ Cout,
    int M, int N, int K) {
  constexpr int LDT = 40;
  __shared__ __align__(16) unsigned short As[2][128 * LDT];
  __shared__ __align__(16) unsigned short Bs[2][128 * LDT];

  const int tid   = threadIdx.x;
  const int lane  = tid & 31;
  const int wv    = tid >> 5;       // 0..7
  const int mw    = wv & 1;         // 2 waves along M
  const int nw    = wv >> 1;        // 4 waves along N
  const int mOff  = mw * 64;
  const int nOff  = nw * 32;
  const int lhalf = lane >> 4;
  const int l16   = lane & 15;

  const int n0 = blockIdx.x * 128;
  const int m0 = blockIdx.y * 128;

  f32x8 acc[4][2] = {};

  // staging: thread -> (row, 16-ushort chunk)
  const int srow  = tid >> 1;   // 0..127
  const int shalf = tid & 1;
  const unsigned aoff0 = (unsigned)(((size_t)(m0 + srow) * K + shalf * 16) * 2);
  const unsigned woff0 = (unsigned)(((size_t)(n0 + srow) * K + shalf * 16) * 2);
  unsigned ldsA[2], ldsB[2];
  ldsA[0] = lds_off(&As[0][srow * LDT + shalf * 16]);
  ldsA[1] = lds_off(&As[1][srow * LDT + shalf * 16]);
  ldsB[0] = lds_off(&Bs[0][srow * LDT + shalf * 16]);
  ldsB[1] = lds_off(&Bs[1][srow * LDT + shalf * 16]);

  // prologue: DMA first K-slab into buffer 0
  async_copy32(ldsA[0], aoff0, A);
  async_copy32(ldsB[0], woff0, W);
  wait_async0();
  __syncthreads();

  int cur = 0;
  for (int k0 = 0; k0 < K; k0 += 32, cur ^= 1) {
    if (k0 + 32 < K) {  // DMA next slab into the other buffer (overlaps WMMA)
      unsigned d = (unsigned)(k0 + 32) * 2;
      async_copy32(ldsA[cur ^ 1], aoff0 + d, A);
      async_copy32(ldsB[cur ^ 1], woff0 + d, W);
    }

    const unsigned short* Ab = As[cur];
    const unsigned short* Bb = Bs[cur];
    Frag af[4], bf_[2];
#pragma unroll
    for (int mi = 0; mi < 4; ++mi) {
      // A 16-bit layout: elem i -> K = i%8 + (i/8)*16 + half*8
      const unsigned short* p = Ab + (mOff + mi * 16 + l16) * LDT + lhalf * 8;
      af[mi].q[0] = *(const uint4*)p;
      af[mi].q[1] = *(const uint4*)(p + 16);
    }
#pragma unroll
    for (int ni = 0; ni < 2; ++ni) {
      // B 16-bit layout: elem i -> K = i%8 + (i/8)*16 + half*8
      const unsigned short* p = Bb + (nOff + ni * 16 + l16) * LDT + lhalf * 8;
      bf_[ni].q[0] = *(const uint4*)p;
      bf_[ni].q[1] = *(const uint4*)(p + 16);
    }
#pragma unroll
    for (int mi = 0; mi < 4; ++mi)
#pragma unroll
      for (int ni = 0; ni < 2; ++ni)
        acc[mi][ni] = __builtin_amdgcn_wmma_f32_16x16x32_bf16(
            false, af[mi].v, false, bf_[ni].v, (short)0, acc[mi][ni], false, false);

    wait_async0();     // next buffer's DMA landed (this wave)
    __syncthreads();   // ... and everyone else's; current buffer now reusable
  }

#pragma unroll
  for (int mi = 0; mi < 4; ++mi)
#pragma unroll
    for (int ni = 0; ni < 2; ++ni) {
      int ncol = n0 + nOff + ni * 16 + l16;
      float bv = bias[ncol];
#pragma unroll
      for (int r = 0; r < 8; ++r) {
        int mrow = m0 + mOff + mi * 16 + r + lhalf * 8;
        float val = acc[mi][ni][r] + bv;
        if (BF16_OUT)
          ((unsigned short*)Cout)[(size_t)mrow * N + ncol] = bf16_rn(val);
        else
          ((float*)Cout)[(size_t)mrow * N + ncol] = val;
      }
    }
}

// ---------------------------------------------------------------------------
// Per-t Hydra attention: 1 block per t, 256 threads = 8 waves (wave w == batch b).
//   qn/kn: L2-normalized q,k in bf16 [128 x 72]
//   vT:    v transposed  bf16 [64 x 136]
//   Ps:    softmax probs bf16 [128 x 136]  (aliases qn/kn region after use)
// ---------------------------------------------------------------------------
__global__ __launch_bounds__(256) void hydra_attn_kernel(
    const unsigned short* __restrict__ qkv,   // [16384 x 3072] bf16, row = t*8+b
    unsigned short* __restrict__ attn) {      // [16384 x 1024] bf16, row = b*T+t
  constexpr int LQ = 72;
  constexpr int LV = 136;
  __shared__ __align__(16) unsigned short smem[2 * 128 * LQ + 64 * LV];  // 54272 B
  unsigned short* qn = smem;                 // 128*72
  unsigned short* kn = smem + 128 * LQ;      // 128*72
  unsigned short* vT = smem + 2 * 128 * LQ;  // 64*136
  unsigned short* Ps = smem;                 // 128*136 = 17408 <= 18432, aliased

  const int t     = blockIdx.x;
  const int tid   = threadIdx.x;
  const int lane  = tid & 31;
  const int wv    = tid >> 5;
  const int lhalf = lane >> 4;
  const int l16   = lane & 15;

  // ---- stage q,k rows with L2 normalization (threads 0..127 -> q, 128..255 -> k)
  {
    int row = tid & 127;
    int isK = tid >> 7;
    int b = row >> 4, h = row & 15;
    const uint4* s4 =
        (const uint4*)(qkv + (size_t)(t * 8 + b) * QKVN + isK * 1024 + h * 64);
    float vals[64];
    float ss = 0.f;
#pragma unroll
    for (int c = 0; c < 8; ++c) {
      union { uint4 q; unsigned short u[8]; } u; u.q = s4[c];
#pragma unroll
      for (int j = 0; j < 8; ++j) {
        float f = bf16_f32(u.u[j]);
        vals[c * 8 + j] = f;
        ss += f * f;
      }
    }
    float inv = rsqrtf(ss);
    unsigned short* dst = (isK ? kn : qn) + row * LQ;
#pragma unroll
    for (int c = 0; c < 64; ++c) dst[c] = bf16_rn(vals[c] * inv);
  }
  // ---- stage v transposed: 2 threads per row, 32 elems each
  {
    int row = tid >> 1;
    int half = tid & 1;
    int b = row >> 4, h = row & 15;
    const uint4* s4 =
        (const uint4*)(qkv + (size_t)(t * 8 + b) * QKVN + 2048 + h * 64 + half * 32);
#pragma unroll
    for (int c4 = 0; c4 < 4; ++c4) {
      union { uint4 q; unsigned short u[8]; } u; u.q = s4[c4];
#pragma unroll
      for (int j = 0; j < 8; ++j)
        vT[(half * 32 + c4 * 8 + j) * LV + row] = u.u[j];
    }
  }
  __syncthreads();

  // ---- scores = qn @ kn^T : wave w owns rows [w*16, w*16+16), all 128 cols
  f32x8 sacc[8] = {};
  for (int k0 = 0; k0 < 64; k0 += 32) {
    Frag af;
    const unsigned short* ap = qn + (wv * 16 + l16) * LQ + k0 + lhalf * 8;
    af.q[0] = *(const uint4*)ap;
    af.q[1] = *(const uint4*)(ap + 16);
#pragma unroll
    for (int ni = 0; ni < 8; ++ni) {
      Frag bfr;
      const unsigned short* bp = kn + (ni * 16 + l16) * LQ + k0 + lhalf * 8;
      bfr.q[0] = *(const uint4*)bp;
      bfr.q[1] = *(const uint4*)(bp + 16);
      sacc[ni] = __builtin_amdgcn_wmma_f32_16x16x32_bf16(
          false, af.v, false, bfr.v, (short)0, sacc[ni], false, false);
    }
  }

  // ---- softmax over cols (scale 1/sqrt(64)=0.125 folded into exp)
  float rmax[8], rsum[8];
#pragma unroll
  for (int r = 0; r < 8; ++r) {
    float m = -1e30f;
#pragma unroll
    for (int ni = 0; ni < 8; ++ni) m = fmaxf(m, sacc[ni][r]);
#pragma unroll
    for (int off = 8; off >= 1; off >>= 1) m = fmaxf(m, __shfl_xor(m, off, 32));
    rmax[r] = m;
  }
#pragma unroll
  for (int r = 0; r < 8; ++r) {
    float s = 0.f;
#pragma unroll
    for (int ni = 0; ni < 8; ++ni) {
      float e = __expf(0.125f * (sacc[ni][r] - rmax[r]));
      sacc[ni][r] = e;
      s += e;
    }
#pragma unroll
    for (int off = 8; off >= 1; off >>= 1) s += __shfl_xor(s, off, 32);
    rsum[r] = s;
  }

  __syncthreads();  // everyone done reading qn/kn before Ps overwrites them
#pragma unroll
  for (int ni = 0; ni < 8; ++ni)
#pragma unroll
    for (int r = 0; r < 8; ++r) {
      int row = wv * 16 + r + lhalf * 8;
      int col = ni * 16 + l16;
      Ps[row * LV + col] = bf16_rn(sacc[ni][r] / rsum[r]);
    }
  __syncthreads();

  // ---- ws = P @ v : wave w -> 16 rows x 64 cols, K=128
  f32x8 oacc[4] = {};
  for (int k0 = 0; k0 < 128; k0 += 32) {
    Frag af;
    const unsigned short* ap = Ps + (wv * 16 + l16) * LV + k0 + lhalf * 8;
    af.q[0] = *(const uint4*)ap;
    af.q[1] = *(const uint4*)(ap + 16);
#pragma unroll
    for (int ni = 0; ni < 4; ++ni) {
      Frag bfr;
      const unsigned short* bp = vT + (ni * 16 + l16) * LV + k0 + lhalf * 8;
      bfr.q[0] = *(const uint4*)bp;
      bfr.q[1] = *(const uint4*)(bp + 16);
      oacc[ni] = __builtin_amdgcn_wmma_f32_16x16x32_bf16(
          false, af.v, false, bfr.v, (short)0, oacc[ni], false, false);
    }
  }

  // rows a = wv*16 + r + half*8 -> b = wv, h = a&15; out row = b*T + t, col h*64+d
#pragma unroll
  for (int ni = 0; ni < 4; ++ni) {
    int d = ni * 16 + l16;
#pragma unroll
    for (int r = 0; r < 8; ++r) {
      int h = r + lhalf * 8;
      attn[((size_t)wv * T_LEN + t) * EMB + h * HD + d] = bf16_rn(oacc[ni][r]);
    }
  }
}

// ---------------------------------------------------------------------------
extern "C" void kernel_launch(void* const* d_in, const int* in_sizes, int n_in,
                              void* d_out, int out_size, void* d_ws, size_t ws_size,
                              hipStream_t stream) {
  (void)in_sizes; (void)n_in; (void)out_size; (void)ws_size;
  const float* query = (const float*)d_in[0];
  // d_in[1]=key, d_in[2]=value: only shapes used by the reference
  const float* w_qkv = (const float*)d_in[3];
  const float* b_qkv = (const float*)d_in[4];
  const float* w_out = (const float*)d_in[5];
  const float* b_out = (const float*)d_in[6];
  float* out = (float*)d_out;

  char* ws = (char*)d_ws;
  unsigned short* qkv_bf  = (unsigned short*)(ws);
  unsigned short* q_bf    = (unsigned short*)(ws + 100663296ULL);
  unsigned short* attn_bf = (unsigned short*)(ws + 134217728ULL);
  unsigned short* wqkv_bf = (unsigned short*)(ws + 167772160ULL);
  unsigned short* wout_bf = (unsigned short*)(ws + 174063616ULL);

  cvt_f32_bf16_kernel<<<2048, 256, 0, stream>>>(query, q_bf, MROWS * EMB / 4);
  cvt_f32_bf16_kernel<<<1024, 256, 0, stream>>>(w_qkv, wqkv_bf, QKVN * EMB / 4);
  cvt_f32_bf16_kernel<<<512,  256, 0, stream>>>(w_out, wout_bf, EMB * EMB / 4);

  // qkv = query @ w_qkv^T + b_qkv   (rows t*8+b)
  gemm_bf16_kernel<true><<<dim3(QKVN / 128, MROWS / 128), 256, 0, stream>>>(
      q_bf, wqkv_bf, b_qkv, qkv_bf, MROWS, QKVN, EMB);

  // per-t normalized attention -> attn (rows b*T+t)
  hydra_attn_kernel<<<T_LEN, 256, 0, stream>>>(qkv_bf, attn_bf);

  // out = attn @ w_out^T + b_out
  gemm_bf16_kernel<false><<<dim3(EMB / 128, MROWS / 128), 256, 0, stream>>>(
      attn_bf, wout_bf, b_out, out, MROWS, EMB, EMB);
}
